// Seq2Seq_88759794139384
// MI455X (gfx1250) — matrix operand order repaired and checked by reading, not compiled, not run on hardware
//
#include <hip/hip_runtime.h>
#include <hip/hip_bf16.h>

// ---------------------------------------------------------------------------
// Seq2Seq transformer forward for MI455X (gfx1250, wave32, WMMA).
// All matmuls run through v_wmma_f32_16x16x32_bf16 with f32 accumulation.
// Activations feeding GEMMs are kept in bf16 (produced in producer epilogues)
// so GEMM inner loops are pure load_b128 + wmma, double-buffered.
// ---------------------------------------------------------------------------

constexpr int kDim   = 512;
constexpr int kHeads = 8;
constexpr int kHd    = 64;          // head dim
constexpr int kFF    = 2048;        // 4*DIM
constexpr int kSeq   = 512;
constexpr int kBatch = 8;
constexpr int kTok   = kBatch * kSeq;   // 4096 rows
constexpr int kVocab = 16384;
constexpr float kEps      = 1e-5f;
constexpr float kSqrtDim  = 22.627416997969522f;   // sqrt(512)
constexpr float kScale    = 0.044194173824159216f; // 1/sqrt(512)  (per reference)
constexpr float kNegInf   = -1.0e30f;

typedef __attribute__((ext_vector_type(16))) __bf16   bf16x16;
typedef __attribute__((ext_vector_type(8)))  float    f32x8;
typedef __attribute__((ext_vector_type(4)))  unsigned u32x4;

union BFrag {              // 32 bytes: one WMMA bf16 operand fragment per lane
  bf16x16        v;
  unsigned short s[16];
  u32x4          q[2];
};

__device__ __forceinline__ unsigned short f2bf(float f) {
  union { float f; unsigned u; } x; x.f = f;
  unsigned u = x.u;
  u += 0x7fffu + ((u >> 16) & 1u);   // round-to-nearest-even
  return (unsigned short)(u >> 16);
}

// ---------------------------------------------------------------------------
// GEMM: C[M,N] = act( A[M,K](bf16) @ W[K,N] + bias ),  W given bf16-transposed
// as Wt[N,K].  One wave per 16x64 tile; double-buffered fragment pipeline.
// Cf (f32) and/or Cbf (bf16) outputs, either may be null.
// ---------------------------------------------------------------------------
__global__ __launch_bounds__(32)
void gemm_bf16_wmma(const unsigned short* __restrict__ Abf,
                    const unsigned short* __restrict__ Wt,
                    const float* __restrict__ bias,
                    float* __restrict__ Cf,
                    unsigned short* __restrict__ Cbf,
                    int M, int N, int K, int gelu)
{
  const int lane = threadIdx.x & 31;
  const int l15  = lane & 15;
  const int lo   = (lane < 16) ? 0 : 8;    // A K-offset == C row offset
  const int kb16 = (lane < 16) ? 0 : 8;    // B K-offset
  const int mtile = blockIdx.x;
  const int nbase = blockIdx.y * 64;

  const unsigned short* arow = Abf + (size_t)(mtile * 16 + l15) * K;
  const unsigned short* wrow = Wt + (size_t)(nbase + l15) * K + kb16;
  const size_t nstride = (size_t)16 * K;

  auto loadA = [&](int kb, BFrag& af) {
    af.q[0] = *(const u32x4*)(arow + kb + lo);
    af.q[1] = *(const u32x4*)(arow + kb + 16 + lo);
  };
  auto loadB = [&](int kb, BFrag* bf) {
#pragma unroll
    for (int t = 0; t < 4; ++t) {
      const u32x4* bp = (const u32x4*)(wrow + kb + (size_t)t * nstride);
      bf[t].q[0] = bp[0];
      bf[t].q[1] = bp[2];
    }
  };

  f32x8 acc[4] = {};
  BFrag a0, a1, b0[4], b1[4];
  loadA(0, a0);
  loadB(0, b0);

  for (int kb = 0; kb < K; kb += 64) {       // K is a multiple of 64
    loadA(kb + 32, a1);
    loadB(kb + 32, b1);
#pragma unroll
    for (int t = 0; t < 4; ++t)
      acc[t] = __builtin_amdgcn_wmma_f32_16x16x32_bf16(
          false, a0.v, false, b0[t].v, (short)0, acc[t], false, false);
    if (kb + 64 < K) {
      loadA(kb + 64, a0);
      loadB(kb + 64, b0);
    }
#pragma unroll
    for (int t = 0; t < 4; ++t)
      acc[t] = __builtin_amdgcn_wmma_f32_16x16x32_bf16(
          false, a1.v, false, b1[t].v, (short)0, acc[t], false, false);
  }

#pragma unroll
  for (int t = 0; t < 4; ++t) {
    const int ncol = nbase + t * 16 + l15;
    const float bv = bias[ncol];
    const size_t rbase = (size_t)(mtile * 16 + lo) * N + ncol;
#pragma unroll
    for (int r = 0; r < 8; ++r) {
      float v = acc[t][r] + bv;
      if (gelu) v = 0.5f * v * (1.0f + erff(v * 0.70710678118654752f));
      if (Cf)  Cf[rbase + (size_t)r * N] = v;
      if (Cbf) Cbf[rbase + (size_t)r * N] = f2bf(v);
    }
  }
}

// ---------------------------------------------------------------------------
// Fused attention: one wave per (batch n, head h, 16-query tile).
//   S = (Q Kh^T) * scale  -> mask -> softmax -> bf16 P (LDS) -> O = P Vh
// Qbf/Kbf: bf16 [tok, DIM]; Vt: bf16 [n, h, d, key]; Obf: bf16 [tok, DIM]
// ---------------------------------------------------------------------------
__global__ __launch_bounds__(32)
void attn_fused_wmma(const unsigned short* __restrict__ Qbf,
                     const unsigned short* __restrict__ Kb,
                     const unsigned short* __restrict__ Vt,
                     const int* __restrict__ keytok,
                     unsigned short* __restrict__ Obf,
                     int causal)
{
  __shared__ __align__(16) float          S[16 * kSeq];   // 32 KB scores
  __shared__ __align__(16) unsigned short P[16 * kSeq];   // 16 KB probs (bf16)
  const int lane = threadIdx.x & 31;
  const int l15  = lane & 15;
  const int lo   = (lane < 16) ? 0 : 8;
  const int kb16 = (lane < 16) ? 0 : 8;
  const int qt = blockIdx.x;   // query tile (0..31)
  const int h  = blockIdx.y;
  const int n  = blockIdx.z;

  // Q fragments for the whole head dim (two K=32 slabs), loaded once
  BFrag qa0, qa1;
  {
    const unsigned short* qp =
        Qbf + (size_t)(n * kSeq + qt * 16 + l15) * kDim + h * kHd;
    qa0.q[0] = *(const u32x4*)(qp + lo);
    qa0.q[1] = *(const u32x4*)(qp + 16 + lo);
    qa1.q[0] = *(const u32x4*)(qp + 32 + lo);
    qa1.q[1] = *(const u32x4*)(qp + 48 + lo);
  }

  // ---- scores ----
  for (int t = 0; t < 32; ++t) {
    const int key = t * 16 + l15;
    const unsigned short* kp =
        Kb + (size_t)(n * kSeq + key) * kDim + h * kHd + kb16;
    BFrag b0, b1;
    b0.q[0] = ((const u32x4*)kp)[0];        b0.q[1] = ((const u32x4*)kp)[2];
    b1.q[0] = ((const u32x4*)(kp + 32))[0]; b1.q[1] = ((const u32x4*)(kp + 32))[2];
    f32x8 acc = {};
    acc = __builtin_amdgcn_wmma_f32_16x16x32_bf16(false, qa0.v, false, b0.v,
                                                  (short)0, acc, false, false);
    acc = __builtin_amdgcn_wmma_f32_16x16x32_bf16(false, qa1.v, false, b1.v,
                                                  (short)0, acc, false, false);
    const bool kvalid = (keytok[n * kSeq + key] != 0);
#pragma unroll
    for (int r = 0; r < 8; ++r) {
      const int qabs = qt * 16 + r + lo;
      const bool ok = kvalid && (!causal || key <= qabs);
      S[(r + lo) * kSeq + key] = ok ? acc[r] * kScale : kNegInf;
    }
  }
  __syncthreads();

  // ---- softmax over each of the 16 rows (2 lanes per row) -> bf16 P ----
  {
    const int row  = l15;
    const int half = lane >> 4;
    const float* sr = &S[row * kSeq + half * 256];
    unsigned short* pr = &P[row * kSeq + half * 256];
    float mx = kNegInf;
    for (int c = 0; c < 256; ++c) mx = fmaxf(mx, sr[c]);
    mx = fmaxf(mx, __shfl_xor(mx, 16, 32));
    float sum = 0.0f;
    for (int c = 0; c < 256; ++c) sum += __expf(sr[c] - mx);
    sum += __shfl_xor(sum, 16, 32);
    const float inv = 1.0f / sum;
    for (int c = 0; c < 256; ++c) pr[c] = f2bf(__expf(sr[c] - mx) * inv);
  }
  __syncthreads();

  // ---- O = P @ Vh  (contraction over 512 keys, 4 d-tiles of 16) ----
  f32x8 oacc[4] = {};
  for (int kb = 0; kb < kSeq; kb += 32) {
    BFrag pa;
    const unsigned short* pr = &P[l15 * kSeq + kb];
    pa.q[0] = *(const u32x4*)(pr + lo);
    pa.q[1] = *(const u32x4*)(pr + 16 + lo);
    const unsigned short* vbase =
        Vt + ((size_t)((n * kHeads + h) * kHd) + l15) * kSeq + kb + kb16;
#pragma unroll
    for (int dt = 0; dt < 4; ++dt) {
      const u32x4* vp = (const u32x4*)(vbase + (size_t)(dt * 16) * kSeq);
      BFrag bf;
      bf.q[0] = vp[0];
      bf.q[1] = vp[2];
      oacc[dt] = __builtin_amdgcn_wmma_f32_16x16x32_bf16(
          false, pa.v, false, bf.v, (short)0, oacc[dt], false, false);
    }
  }
  {
    unsigned short* orow =
        Obf + (size_t)(n * kSeq + qt * 16 + lo) * kDim + h * kHd + l15;
#pragma unroll
    for (int dt = 0; dt < 4; ++dt)
#pragma unroll
      for (int r = 0; r < 8; ++r)
        orow[(size_t)r * kDim + dt * 16] = f2bf(oacc[dt][r]);
  }
}

// ---------------------------------------------------------------------------
// LayerNorm kernels (one 256-thread block per 512-wide row); write f32 + bf16
// ---------------------------------------------------------------------------
__device__ __forceinline__ float block_reduce_sum(float v, float* red) {
  const int tid = threadIdx.x;
  red[tid] = v;
  __syncthreads();
  for (int s = 128; s > 0; s >>= 1) {
    if (tid < s) red[tid] += red[tid + s];
    __syncthreads();
  }
  float r = red[0];
  __syncthreads();
  return r;
}

__global__ __launch_bounds__(256)
void embed_ln_k(const int* __restrict__ toks, const float* __restrict__ tokT,
                const float* __restrict__ posT, const float* __restrict__ g,
                const float* __restrict__ b,
                float* __restrict__ outf, unsigned short* __restrict__ outbf)
{
  __shared__ float red[256];
  const int row = blockIdx.x;
  const int tid = threadIdx.x;
  const int pos = row & (kSeq - 1);
  const int tk  = toks[row];
  float e0 = tokT[(size_t)tk * kDim + tid] * kSqrtDim + posT[(size_t)pos * kDim + tid];
  float e1 = tokT[(size_t)tk * kDim + tid + 256] * kSqrtDim +
             posT[(size_t)pos * kDim + tid + 256];
  const float mu  = block_reduce_sum(e0 + e1, red) * (1.0f / kDim);
  const float d0 = e0 - mu, d1 = e1 - mu;
  const float var = block_reduce_sum(d0 * d0 + d1 * d1, red) * (1.0f / kDim);
  const float rs  = rsqrtf(var + kEps);
  const float v0 = d0 * rs * g[tid] + b[tid];
  const float v1 = d1 * rs * g[tid + 256] + b[tid + 256];
  outf[(size_t)row * kDim + tid]        = v0;
  outf[(size_t)row * kDim + tid + 256]  = v1;
  outbf[(size_t)row * kDim + tid]       = f2bf(v0);
  outbf[(size_t)row * kDim + tid + 256] = f2bf(v1);
}

__global__ __launch_bounds__(256)
void add_ln_k(const float* __restrict__ a, const float* __restrict__ r,
              const float* __restrict__ g, const float* __restrict__ b,
              float* __restrict__ outf, unsigned short* __restrict__ outbf)
{
  __shared__ float red[256];
  const int row = blockIdx.x;
  const int tid = threadIdx.x;
  const size_t base = (size_t)row * kDim;
  float e0 = a[base + tid] + r[base + tid];
  float e1 = a[base + tid + 256] + r[base + tid + 256];
  const float mu  = block_reduce_sum(e0 + e1, red) * (1.0f / kDim);
  const float d0 = e0 - mu, d1 = e1 - mu;
  const float var = block_reduce_sum(d0 * d0 + d1 * d1, red) * (1.0f / kDim);
  const float rs  = rsqrtf(var + kEps);
  const float v0 = d0 * rs * g[tid] + b[tid];
  const float v1 = d1 * rs * g[tid + 256] + b[tid + 256];
  outf[base + tid]        = v0;
  outf[base + tid + 256]  = v1;
  outbf[base + tid]       = f2bf(v0);
  outbf[base + tid + 256] = f2bf(v1);
}

// ---------------------------------------------------------------------------
// Conversion kernels
// ---------------------------------------------------------------------------
__global__ __launch_bounds__(256)
void wcast_T(const float* __restrict__ W, unsigned short* __restrict__ Wt,
             int K, int N)   // W[K,N] f32 -> Wt[N,K] bf16
{
  int idx = blockIdx.x * 256 + threadIdx.x;
  if (idx >= K * N) return;
  int kk = idx / N, nn = idx - kk * N;
  Wt[(size_t)nn * K + kk] = f2bf(W[idx]);
}

__global__ __launch_bounds__(256)
void vtrans_bf(const unsigned short* __restrict__ Vrow,
               unsigned short* __restrict__ Vt)
{ // Vrow bf16 [tok=n*512+key, col=h*64+d] -> Vt bf16 [n,h,d,key]
  int idx = blockIdx.x * 256 + threadIdx.x;
  if (idx >= kTok * kDim) return;
  int tok = idx >> 9, c = idx & 511;
  int n = tok >> 9, key = tok & 511;
  int h = c >> 6, d = c & 63;
  Vt[(((size_t)(n * kHeads + h) * kHd) + d) * kSeq + key] = Vrow[idx];
}

// ---------------------------------------------------------------------------
// Host orchestration
// ---------------------------------------------------------------------------
struct AttnP  { const float *qw,*qb,*kw,*kb,*vw,*vb,*ow,*ob; };
struct LNP    { const float *g,*b; };
struct BlockP { AttnP attn; const float *f1w,*f1b,*f2w,*f2b; LNP ln1, ln2; };

static inline const float* FP(void* const* d, int i) { return (const float*)d[i]; }

static AttnP attnAt(void* const* d, int i) {   // sorted leaves: k.b k.w o.b o.w q.b q.w v.b v.w
  AttnP a;
  a.kb = FP(d, i + 0); a.kw = FP(d, i + 1);
  a.ob = FP(d, i + 2); a.ow = FP(d, i + 3);
  a.qb = FP(d, i + 4); a.qw = FP(d, i + 5);
  a.vb = FP(d, i + 6); a.vw = FP(d, i + 7);
  return a;
}
static BlockP blockAt(void* const* d, int i) { // attn(8), ff1.b/w, ff2.b/w, ln1.b/g, ln2.b/g
  BlockP p;
  p.attn = attnAt(d, i);
  p.f1b = FP(d, i + 8);  p.f1w = FP(d, i + 9);
  p.f2b = FP(d, i + 10); p.f2w = FP(d, i + 11);
  p.ln1.b = FP(d, i + 12); p.ln1.g = FP(d, i + 13);
  p.ln2.b = FP(d, i + 14); p.ln2.g = FP(d, i + 15);
  return p;
}

struct WSBuf {
  unsigned short *WB;                      // bf16 transposed weights (scratch)
  unsigned short *Qbf, *Kbf, *Vrow, *Vt, *Obf;
  unsigned short *T2bf, *Hbf, *Xbf, *Ybf;
  float *T1f, *T2f, *Xf, *Yf;
};

static void run_linear(const unsigned short* Abf, const float* W, const float* bias,
                       float* Cf, unsigned short* Cbf,
                       int M, int N, int K, int gelu, WSBuf& ws, hipStream_t s)
{
  const int total = K * N;
  wcast_T<<<(total + 255) / 256, 256, 0, s>>>(W, ws.WB, K, N);
  dim3 grid(M / 16, N / 64);
  gemm_bf16_wmma<<<grid, 32, 0, s>>>(Abf, ws.WB, bias, Cf, Cbf, M, N, K, gelu);
}

static void run_mha(const AttnP& p, const unsigned short* qin_bf,
                    const unsigned short* kv_bf, const int* keytok, int causal,
                    float* outf, WSBuf& ws, hipStream_t s)
{
  run_linear(qin_bf, p.qw, p.qb, nullptr, ws.Qbf, kTok, kDim, kDim, 0, ws, s);
  run_linear(kv_bf,  p.kw, p.kb, nullptr, ws.Kbf, kTok, kDim, kDim, 0, ws, s);
  run_linear(kv_bf,  p.vw, p.vb, nullptr, ws.Vrow, kTok, kDim, kDim, 0, ws, s);
  vtrans_bf<<<(kTok * kDim) / 256, 256, 0, s>>>(ws.Vrow, ws.Vt);
  dim3 grid(kSeq / 16, kHeads, kBatch);
  attn_fused_wmma<<<grid, 32, 0, s>>>(ws.Qbf, ws.Kbf, ws.Vt, keytok, ws.Obf, causal);
  run_linear(ws.Obf, p.ow, p.ob, outf, nullptr, kTok, kDim, kDim, 0, ws, s);
}

static void run_block(const BlockP& p, float* xf, unsigned short* xbf,
                      const unsigned short* kvbf, const int* keytok, int causal,
                      WSBuf& ws, hipStream_t s)
{
  run_mha(p.attn, xbf, kvbf, keytok, causal, ws.T1f, ws, s);
  add_ln_k<<<kTok, 256, 0, s>>>(ws.T1f, xf, p.ln1.g, p.ln1.b, ws.T2f, ws.T2bf);
  run_linear(ws.T2bf, p.f1w, p.f1b, nullptr, ws.Hbf, kTok, kFF, kDim, 1, ws, s);
  run_linear(ws.Hbf, p.f2w, p.f2b, ws.T1f, nullptr, kTok, kDim, kFF, 0, ws, s);
  add_ln_k<<<kTok, 256, 0, s>>>(ws.T1f, ws.T2f, p.ln2.g, p.ln2.b, xf, xbf);
}

extern "C" void kernel_launch(void* const* d_in, const int* in_sizes, int n_in,
                              void* d_out, int out_size, void* d_ws, size_t ws_size,
                              hipStream_t stream)
{
  (void)in_sizes; (void)n_in; (void)out_size; (void)ws_size;

  // ---- carve workspace (~90 MB) ----
  char* base = (char*)d_ws;
  size_t off = 0;
  auto carve = [&](size_t bytes) -> void* {
    void* p = base + off;
    off += (bytes + 255) & ~(size_t)255;
    return p;
  };
  WSBuf ws;
  ws.WB   = (unsigned short*)carve((size_t)kDim * kVocab * 2);  // largest Wt
  ws.Qbf  = (unsigned short*)carve((size_t)kTok * kDim * 2);
  ws.Kbf  = (unsigned short*)carve((size_t)kTok * kDim * 2);
  ws.Vrow = (unsigned short*)carve((size_t)kTok * kDim * 2);
  ws.Vt   = (unsigned short*)carve((size_t)kTok * kDim * 2);
  ws.Obf  = (unsigned short*)carve((size_t)kTok * kDim * 2);
  ws.T2bf = (unsigned short*)carve((size_t)kTok * kDim * 2);
  ws.Hbf  = (unsigned short*)carve((size_t)kTok * kFF * 2);
  ws.Xbf  = (unsigned short*)carve((size_t)kTok * kDim * 2);
  ws.Ybf  = (unsigned short*)carve((size_t)kTok * kDim * 2);
  ws.T1f  = (float*)carve((size_t)kTok * kDim * 4);
  ws.T2f  = (float*)carve((size_t)kTok * kDim * 4);
  ws.Xf   = (float*)carve((size_t)kTok * kDim * 4);
  ws.Yf   = (float*)carve((size_t)kTok * kDim * 4);

  // ---- leaf index map (jax tree_flatten order: dict keys sorted) ----
  // 0..47 dec_cross[0..2] (16 ea) | 48..51 dec_emb(ln.b,ln.g,pos,tok)
  // 52..81 dec_self[0..2] (10 ea) | 82..85 enc_emb | 86..133 enc_layers
  // 134 fc.b 135 fc.w | 136 src 137 tgt
  const int* src = (const int*)d_in[136];
  const int* tgt = (const int*)d_in[137];

  // ---- encoder ----
  embed_ln_k<<<kTok, 256, 0, stream>>>(src, FP(d_in, 85), FP(d_in, 84),
                                       FP(d_in, 83), FP(d_in, 82), ws.Xf, ws.Xbf);
  for (int l = 0; l < 3; ++l) {
    BlockP p = blockAt(d_in, 86 + 16 * l);
    run_block(p, ws.Xf, ws.Xbf, ws.Xbf, src, 0, ws, stream);
  }

  // ---- decoder ----
  embed_ln_k<<<kTok, 256, 0, stream>>>(tgt, FP(d_in, 51), FP(d_in, 50),
                                       FP(d_in, 49), FP(d_in, 48), ws.Yf, ws.Ybf);
  for (int l = 0; l < 3; ++l) {
    AttnP sp = attnAt(d_in, 52 + 10 * l);
    const float* s_lnb = FP(d_in, 52 + 10 * l + 8);
    const float* s_lng = FP(d_in, 52 + 10 * l + 9);
    run_mha(sp, ws.Ybf, ws.Ybf, tgt, /*causal=*/1, ws.T1f, ws, stream);
    add_ln_k<<<kTok, 256, 0, stream>>>(ws.T1f, ws.Yf, s_lng, s_lnb, ws.Yf, ws.Ybf);
    BlockP cp = blockAt(d_in, 16 * l);
    run_block(cp, ws.Yf, ws.Ybf, ws.Xbf, src, /*causal=*/0, ws, stream);
  }

  // ---- output projection ----
  run_linear(ws.Ybf, FP(d_in, 135), FP(d_in, 134), (float*)d_out, nullptr,
             kTok, kVocab, kDim, 0, ws, stream);
}
